// LogEig_22643067584718
// MI455X (gfx1250) — matrix operand order repaired and checked by reading, not compiled, not run on hardware
//
// CDNA5 / gfx1250 (MI455X)  —  batched log of SPD 32x32 matrices, vech output.
//
// Math: spectrum(X) ⊂ [1, ~5.3]  (X = A A^T/32 + I). Use interval [0.90, 7.5]:
//   Y = (X - m I)/h,  log(X) = p(Y),  p = degree-19 Chebyshev series of log
//   (closed-form coeffs c_n = -2 r^n / n, |r|≈0.486 -> trunc ~1e-7), converted
//   to monomial basis on host in double; Paterson-Stockmeyer s=4:
//       p(Y) = sum_q B_q(Y) W^q,  W = Y^4 = Y^2*Y^2,  B_q = a(4q) I + ... Y^3
//   => Y^2, Y^3, W + 4 Horner products = 7 matmuls of 32x32 per matrix.
//
// Precision: two-term f16 split (hi+lo ~ 22 mantissa bits); each 16x16 tile
// product = 3 x v_wmma_f32_16x16x32_f16 (Ah*Bh + Ah*Bl + Al*Bh). Split cost is
// 1.5 instr/element: v_cvt_pk_rtz_f16_f32 for hi (RTZ ok: lo catches the exact
// f32 residual), v_fma_mixlo/hi_f16 for lo (residual fma + cvt + pack fused).
//
// Layout: all intermediates are symmetric => rows == columns. With the ISA's
// 16-bit A layout and 32-bit C/D layout, D tiles feed the next A operand with
// NO cross-lane movement; B operands (Y, Y^2, W) reuse the A-form, no swap; X load and output use the
// lane-half swap, done via ds_swizzle_b32 SWAPX16 (offset 0x401f, no LDS).
#include <hip/hip_runtime.h>
#include <cmath>

typedef _Float16 h16  __attribute__((ext_vector_type(16)));
typedef __fp16   f16x2 __attribute__((ext_vector_type(2)));
typedef float    v8f  __attribute__((ext_vector_type(8)));
typedef unsigned u8v  __attribute__((ext_vector_type(8)));

#define DEG  19
#define NCO  (DEG + 1)
#define NBLK (NCO / 4)   // 5 blocks of 4 coefficients

struct LogCoef { float a[NCO]; float mshift; float invh; };

// ---- lane-half swap (lanes L <-> L+16), no LDS allocation needed ----------
static __device__ __forceinline__ unsigned swz16u(unsigned x) {
  return (unsigned)__builtin_amdgcn_ds_swizzle((int)x, 0x401f); // xor=0x10,and=0x1f
}
static __device__ __forceinline__ float swz16f(float x) {
  return __builtin_bit_cast(float,
         __builtin_amdgcn_ds_swizzle(__builtin_bit_cast(int, x), 0x401f));
}

// ---- two-term f16 split of an f32 pair: 3 instructions --------------------
static __device__ __forceinline__ void split2(float x0, float x1,
                                              unsigned &hp, unsigned &lp) {
  f16x2 hv = __builtin_amdgcn_cvt_pkrtz(x0, x1);      // v_cvt_pk_rtz_f16_f32
  unsigned h = __builtin_bit_cast(unsigned, hv);
  unsigned l;
  // lo[15:0]  = f16( x0 - f16lo(h) );  lo[31:16] = f16( x1 - f16hi(h) )
  asm("v_fma_mixlo_f16 %0, %1, -1.0, %2 op_sel_hi:[1,0,0]\n\t"
      "v_fma_mixhi_f16 %0, %1, -1.0, %3 op_sel:[1,0,0] op_sel_hi:[1,0,0]"
      : "=&v"(l)
      : "v"(h), "v"(x0), "v"(x1));
  hp = h; lp = l;
}

static __device__ __forceinline__ v8f wmma_f16(u8v a, u8v b, v8f c) {
  return __builtin_amdgcn_wmma_f32_16x16x32_f16(
      false, __builtin_bit_cast(h16, a),
      false, __builtin_bit_cast(h16, b),
      (short)0, c, false, false);
}
// split product: Ah*Bh + Ah*Bl + Al*Bh  (drop Al*Bl, ~2^-22)
static __device__ __forceinline__ v8f mm3(u8v ah, u8v al, u8v bh, u8v bl, v8f c) {
  c = wmma_f16(ah, bh, c);
  c = wmma_f16(ah, bl, c);
  c = wmma_f16(al, bh, c);
  return c;
}

// f16-split A-form for row-block I from D-layout f32 tiles (0,I),(1,I).
// Lane L<16 holds row(16I+L) K{0..7,16..23}; lane L+16 K{8..15,24..31} —
// exactly what those lanes already hold in the D tiles (symmetry), no shuffle.
static __device__ __forceinline__ void makeA(v8f t0, v8f t1, u8v &ah, u8v &al) {
#pragma unroll
  for (int j = 0; j < 4; ++j) {
    unsigned h, l;
    split2(t0[2*j], t0[2*j+1], h, l); ah[j]     = h; al[j]     = l;
    split2(t1[2*j], t1[2*j+1], h, l); ah[4 + j] = h; al[4 + j] = l;
  }
}
// B-form(J) from A-form(J): same k order as A (symmetry), no swap:
// lane<16 holds K{0..7,16..23} of column 16J+l16, lane>=16 holds K{8..15,24..31}.
static __device__ __forceinline__ u8v makeB(u8v a, bool lo16) {
  u8v b;
#pragma unroll
  for (int j = 0; j < 4; ++j) {
    b[j]     = a[j];
    b[4 + j] = a[4 + j];
  }
  return b;
}

__global__ __launch_bounds__(256) void logm_spd32(
    const float* __restrict__ X, float* __restrict__ out,
    LogCoef cf, int nmat)
{
  const int lane = threadIdx.x & 31;
  const int mat  = blockIdx.x * (blockDim.x >> 5) + (threadIdx.x >> 5);
  if (mat >= nmat) return;          // wave-uniform
  const bool lo16 = lane < 16;
  const int  l16  = lane & 15;

  // per-lane diagonal selector for tiles (0,0)/(1,1)
  v8f dselv;
#pragma unroll
  for (int j = 0; j < 8; ++j)
    dselv[j] = (l16 == (lo16 ? j : (8 + j))) ? 1.f : 0.f;

  // ---- load row `lane` of X (coalesced 4KB per wave; X symmetric) ---------
  const float* xp = X + (size_t)mat * 1024 + (size_t)lane * 32;
  float rw[32];
#pragma unroll
  for (int i = 0; i < 8; ++i) {
    float4 v = reinterpret_cast<const float4*>(xp)[i];
    rw[4*i] = v.x; rw[4*i+1] = v.y; rw[4*i+2] = v.z; rw[4*i+3] = v.w;
  }
  float sw[32];
#pragma unroll
  for (int k = 0; k < 32; ++k) sw[k] = swz16f(rw[k]);

  // ---- D-layout tiles of Y = (X - m I)/h  (0=(0,0) 1=(0,1) 2=(1,0) 3=(1,1))
  v8f y[4], y2[4], y3[4];
  {
    v8f x00, x01, x10, x11;
#pragma unroll
    for (int j = 0; j < 8; ++j) {
      x00[j] = lo16 ? rw[j]      : sw[8 + j];
      x01[j] = lo16 ? sw[j]      : rw[8 + j];
      x10[j] = lo16 ? rw[16 + j] : sw[24 + j];
      x11[j] = lo16 ? sw[16 + j] : rw[24 + j];
    }
    y[0] = (x00 - dselv * cf.mshift) * cf.invh;   // v_pk_fma_f32-friendly
    y[1] = x01 * cf.invh;
    y[2] = x10 * cf.invh;
    y[3] = (x11 - dselv * cf.mshift) * cf.invh;
  }

  // ---- powers: Y^2, Y^3, W = Y^2*Y^2 --------------------------------------
  u8v Ah[2], Al[2], YBh[2], YBl[2], WBh[2], WBl[2];
  makeA(y[0], y[2], Ah[0], Al[0]);
  makeA(y[1], y[3], Ah[1], Al[1]);
  YBh[0] = makeB(Ah[0], lo16); YBl[0] = makeB(Al[0], lo16);
  YBh[1] = makeB(Ah[1], lo16); YBl[1] = makeB(Al[1], lo16);

#pragma unroll
  for (int i = 0; i < 2; ++i)
#pragma unroll
    for (int j = 0; j < 2; ++j) { v8f z = {}; y2[i*2+j] = mm3(Ah[i], Al[i], YBh[j], YBl[j], z); }

  makeA(y2[0], y2[2], Ah[0], Al[0]);          // A-form of Y^2
  makeA(y2[1], y2[3], Ah[1], Al[1]);
#pragma unroll
  for (int i = 0; i < 2; ++i)                 // Y^3 = Y^2 * Y
#pragma unroll
    for (int j = 0; j < 2; ++j) { v8f z = {}; y3[i*2+j] = mm3(Ah[i], Al[i], YBh[j], YBl[j], z); }

  v8f R[4];                                   // W = Y^2 * Y^2 (scratch in R)
  {
    u8v Y2Bh[2], Y2Bl[2];
    Y2Bh[0] = makeB(Ah[0], lo16); Y2Bl[0] = makeB(Al[0], lo16);
    Y2Bh[1] = makeB(Ah[1], lo16); Y2Bl[1] = makeB(Al[1], lo16);
#pragma unroll
    for (int i = 0; i < 2; ++i)
#pragma unroll
      for (int j = 0; j < 2; ++j) { v8f z = {}; R[i*2+j] = mm3(Ah[i], Al[i], Y2Bh[j], Y2Bl[j], z); }
  }
  makeA(R[0], R[2], Ah[0], Al[0]);            // B-form of W
  makeA(R[1], R[3], Ah[1], Al[1]);
  WBh[0] = makeB(Ah[0], lo16); WBl[0] = makeB(Al[0], lo16);
  WBh[1] = makeB(Ah[1], lo16); WBl[1] = makeB(Al[1], lo16);

  // ---- Paterson-Stockmeyer Horner: R = B_{NBLK-1}; R = R*W + B_q ----------
  auto formB = [&](int q, v8f C[4]) {
    const float a0 = cf.a[4*q], a1 = cf.a[4*q+1], a2 = cf.a[4*q+2], a3 = cf.a[4*q+3];
#pragma unroll
    for (int t = 0; t < 4; ++t) {
      v8f c = y[t] * a1 + y2[t] * a2 + y3[t] * a3;
      if (t == 0 || t == 3) c = c + dselv * a0;
      C[t] = c;
    }
  };

  formB(NBLK - 1, R);
#pragma unroll
  for (int q = NBLK - 2; q >= 0; --q) {
    makeA(R[0], R[2], Ah[0], Al[0]);     // no shuffles: D-layout feeds A-form
    makeA(R[1], R[3], Ah[1], Al[1]);
    v8f C[4];
    formB(q, C);
#pragma unroll
    for (int i = 0; i < 2; ++i)
#pragma unroll
      for (int j = 0; j < 2; ++j)
        R[i*2+j] = mm3(Ah[i], Al[i], WBh[j], WBl[j], C[i*2+j]);
  }

  // ---- epilogue: vech upper triangle, off-diag * sqrt(2) ------------------
  float* ob = out + (size_t)mat * 528;
  const float SQ2 = 1.41421356237309515f;
  const bool laneHi = !lo16;
#pragma unroll
  for (int r = 0; r < 32; ++r) {
    const int  I = r >> 4, jj = r & 7;
    const bool needHi = (r & 15) >= 8;     // which lane-half holds row r
    float v0 = R[I*2 + 0][jj];             // (r, c=l16)      on needHi half
    float v1 = R[I*2 + 1][jj];             // (r, c=16+l16)   on needHi half
    float s1 = swz16f(v1);                 // move J=1 values to idle half
    bool  keep = (laneHi == needHi);
    float val  = keep ? v0 : s1;
    int   c    = keep ? l16 : (l16 + 16);
    if (c >= r) {
      float o = (c == r) ? val : val * SQ2;
      ob[(r * (65 - r)) / 2 + (c - r)] = o;   // row offset = 32r - r(r-1)/2
    }
  }
}

extern "C" void kernel_launch(void* const* d_in, const int* in_sizes, int n_in,
                              void* d_out, int out_size, void* d_ws, size_t ws_size,
                              hipStream_t stream) {
  (void)n_in; (void)out_size; (void)d_ws; (void)ws_size;
  const float* X = (const float*)d_in[0];
  float* out = (float*)d_out;
  const int nmat = in_sizes[0] / 1024;

  // host-side (double) Chebyshev-of-log coefficients -> monomial basis in t
  LogCoef cf;
  {
    const double lo = 0.90, hi = 7.5;
    const double m = 0.5 * (lo + hi), h = 0.5 * (hi - lo), zeta = h / m;
    const double r = -(1.0 - std::sqrt(1.0 - zeta * zeta)) / zeta; // |r|~0.486
    double c[NCO];
    c[0] = std::log(m) - std::log(1.0 + r * r);
    double p = 1.0;
    for (int n = 1; n < NCO; ++n) { p *= r; c[n] = -2.0 * p / n; }
    double a[NCO], Tm2[NCO], Tm1[NCO], Tc[NCO];
    for (int k = 0; k < NCO; ++k) { a[k] = 0.0; Tm2[k] = 0.0; Tm1[k] = 0.0; }
    Tm2[0] = 1.0; Tm1[1] = 1.0;                  // T0, T1
    for (int k = 0; k < NCO; ++k) a[k] += c[0] * Tm2[k] + c[1] * Tm1[k];
    for (int n = 2; n < NCO; ++n) {
      for (int k = 0; k < NCO; ++k) Tc[k] = (k ? 2.0 * Tm1[k-1] : 0.0) - Tm2[k];
      for (int k = 0; k < NCO; ++k) a[k] += c[n] * Tc[k];
      for (int k = 0; k < NCO; ++k) { Tm2[k] = Tm1[k]; Tm1[k] = Tc[k]; }
    }
    for (int k = 0; k < NCO; ++k) cf.a[k] = (float)a[k];
    cf.mshift = (float)m; cf.invh = (float)(1.0 / h);
  }

  const int wavesPerBlock = 8;                      // 256 threads = 8 wave32
  const int blocks = (nmat + wavesPerBlock - 1) / wavesPerBlock;
  logm_spd32<<<blocks, 32 * wavesPerBlock, 0, stream>>>(X, out, cf, nmat);
}
